// Ours_9663676416077
// MI455X (gfx1250) — matrix operand order repaired and checked by reading, not compiled, not run on hardware
//
#include <hip/hip_runtime.h>
#include <math.h>

// ---------------------------------------------------------------------------
// Model dims
// ---------------------------------------------------------------------------
#define NN   4096
#define TT   256
#define DLL  768
#define DAA  74
#define DVV  47
#define KAP  96   // DA padded to multiple of 32
#define KVP  64   // DV padded to multiple of 32

typedef __attribute__((ext_vector_type(16))) __bf16 v16bf;
typedef __attribute__((ext_vector_type(8)))  float  v8f;

union FragBF { v16bf v; uint4 q[2]; };

// A-matrix fragment (16x32 bf16, row-major source with given element stride).
// Per ISA 7.12.2: lane<16 -> K = 0..7 & 16..23 ; lane>=16 -> K = 8..15 & 24..31
__device__ __forceinline__ v16bf load_a16(const __bf16* base, int stride, int k0, int lane) {
    int m  = lane & 15;
    int kb = k0 + ((lane >> 4) ? 8 : 0);
    const __bf16* p = base + m * stride + kb;
    FragBF f;
    f.q[0] = *(const uint4*)(p);
    f.q[1] = *(const uint4*)(p + 16);
    return f.v;
}

// B-matrix fragment (32x16 bf16). Source W is row-major [ncols][stride] with
// B[k][n] = W[n*stride + k]. lane<16 -> K = 0..7 & 16..23 ; lane>=16 -> K = 8..15 & 24..31
__device__ __forceinline__ v16bf load_b16(const __bf16* w, int stride, int n0, int k0, int lane) {
    int n  = n0 + (lane & 15);
    int kb = k0 + ((lane >> 4) ? 8 : 0);
    const __bf16* p = w + n * stride + kb;
    FragBF f;
    f.q[0] = *(const uint4*)(p);
    f.q[1] = *(const uint4*)(p + 16);
    return f.v;
}

#define WMMA_BF16(a, b, c) \
    __builtin_amdgcn_wmma_f32_16x16x32_bf16(false, (a), false, (b), (short)0, (c), false, false)

// fast hardware rcp/rsq (single v_rcp_f32 / v_rsq_f32, no NR refinement)
__device__ __forceinline__ float frcp(float x)  { return __builtin_amdgcn_rcpf(x); }
__device__ __forceinline__ float frsq(float x)  { return __builtin_amdgcn_rsqf(x); }
__device__ __forceinline__ float sigm(float x)  { return frcp(1.f + __expf(-x)); }
__device__ __forceinline__ float ftanh(float x) { return 1.f - 2.f * frcp(1.f + __expf(2.f * x)); }

// ---------------------------------------------------------------------------
// Workspace layout (bytes)
// ---------------------------------------------------------------------------
#define OFF_XBUF   0u                    // float [4096*96]
#define OFF_LW1    1572864u              // bf16 [128*768]
#define OFF_LW2    1769472u              // bf16 [32*128]
#define OFF_AWIH   1777664u              // bf16 [128*96] (zero padded)
#define OFF_AWHH   1802240u              // bf16 [128*32]
#define OFF_VWIH   1810432u              // bf16 [128*64] (zero padded)
#define OFF_VWHH   1826816u              // bf16 [128*32]
#define OFF_AWO    1835008u              // bf16 [32*32]
#define OFF_VWO    1837056u              // bf16 [32*32]
#define OFF_ABIAS  1839104u              // float[128]  (abih+abhh)
#define OFF_VBIAS  1839616u              // float[128]

// Shared arena: max(text 28672, lstm-audio 17664)
#define SMEM_BYTES 28672

// ---------------------------------------------------------------------------
// Prep: convert weights to bf16 (with K padding) into workspace
// ---------------------------------------------------------------------------
__global__ __launch_bounds__(256) void prep_kernel(
    const float* __restrict__ lW1, const float* __restrict__ lW2,
    const float* __restrict__ aWih, const float* __restrict__ aWhh,
    const float* __restrict__ abih, const float* __restrict__ abhh,
    const float* __restrict__ vWih, const float* __restrict__ vWhh,
    const float* __restrict__ vbih, const float* __restrict__ vbhh,
    const float* __restrict__ aWo, const float* __restrict__ vWo,
    char* __restrict__ ws)
{
    __bf16* lw1  = (__bf16*)(ws + OFF_LW1);
    __bf16* lw2  = (__bf16*)(ws + OFF_LW2);
    __bf16* awih = (__bf16*)(ws + OFF_AWIH);
    __bf16* awhh = (__bf16*)(ws + OFF_AWHH);
    __bf16* vwih = (__bf16*)(ws + OFF_VWIH);
    __bf16* vwhh = (__bf16*)(ws + OFF_VWHH);
    __bf16* awo  = (__bf16*)(ws + OFF_AWO);
    __bf16* vwo  = (__bf16*)(ws + OFF_VWO);
    float*  abia = (float*)(ws + OFF_ABIAS);
    float*  vbia = (float*)(ws + OFF_VBIAS);

    int g = blockIdx.x * blockDim.x + threadIdx.x;
    int gs = gridDim.x * blockDim.x;
    for (int i = g; i < 128 * 768; i += gs) lw1[i] = (__bf16)lW1[i];
    for (int i = g; i < 32 * 128;  i += gs) lw2[i] = (__bf16)lW2[i];
    for (int i = g; i < 128 * KAP; i += gs) {
        int n = i / KAP, k = i % KAP;
        awih[i] = (k < DAA) ? (__bf16)aWih[n * DAA + k] : (__bf16)0.f;
    }
    for (int i = g; i < 128 * 32;  i += gs) awhh[i] = (__bf16)aWhh[i];
    for (int i = g; i < 128 * KVP; i += gs) {
        int n = i / KVP, k = i % KVP;
        vwih[i] = (k < DVV) ? (__bf16)vWih[n * DVV + k] : (__bf16)0.f;
    }
    for (int i = g; i < 128 * 32;  i += gs) vwhh[i] = (__bf16)vWhh[i];
    for (int i = g; i < 32 * 32;   i += gs) { awo[i] = (__bf16)aWo[i]; vwo[i] = (__bf16)vWo[i]; }
    for (int i = g; i < 128;       i += gs) { abia[i] = abih[i] + abhh[i]; vbia[i] = vbih[i] + vbhh[i]; }
}

// ---------------------------------------------------------------------------
// Text path: lf = relu(relu(l @ lW1^T + lb1) @ lW2^T + lb2)   -> xbuf[:, 0:32]
// ---------------------------------------------------------------------------
__device__ void text_tile(const float* __restrict__ l, const float* __restrict__ lb1,
                          const float* __restrict__ lb2, const char* __restrict__ ws,
                          float* __restrict__ xbuf, char* smem)
{
    const __bf16* lw1 = (const __bf16*)(ws + OFF_LW1);
    const __bf16* lw2 = (const __bf16*)(ws + OFF_LW2);

    __bf16* lt = (__bf16*)smem;                       // [16][768]
    __bf16* h1 = (__bf16*)(smem + 16 * DLL * 2);      // [16][128]

    int tid = threadIdx.x, lane = tid & 31, wave = tid >> 5;
    int n0 = blockIdx.x * 16;

    const float* src = l + (size_t)n0 * DLL;
    for (int i = tid; i < 16 * DLL; i += 256) lt[i] = (__bf16)src[i];
    __syncthreads();

    v8f acc = {};
    int c0 = wave * 16;
    for (int ks = 0; ks < DLL; ks += 32) {
        v16bf a = load_a16(lt, DLL, ks, lane);
        v16bf b = load_b16(lw1, DLL, c0, ks, lane);
        acc = WMMA_BF16(a, b, acc);
    }
    {
        int n = c0 + (lane & 15), mb = (lane >> 4) * 8;
        float bias = lb1[n];
        #pragma unroll
        for (int r = 0; r < 8; ++r)
            h1[(mb + r) * 128 + n] = (__bf16)fmaxf(acc[r] + bias, 0.f);
    }
    __syncthreads();

    if (wave < 2) {
        v8f a2 = {};
        int cc0 = wave * 16;
        #pragma unroll
        for (int ks = 0; ks < 128; ks += 32) {
            v16bf a = load_a16(h1, 128, ks, lane);
            v16bf b = load_b16(lw2, 128, cc0, ks, lane);
            a2 = WMMA_BF16(a, b, a2);
        }
        int n = cc0 + (lane & 15), mb = (lane >> 4) * 8;
        float bias = lb2[n];
        #pragma unroll
        for (int r = 0; r < 8; ++r)
            xbuf[(size_t)(n0 + mb + r) * 96 + n] = fmaxf(a2[r] + bias, 0.f);
    }
}

// ---------------------------------------------------------------------------
// LSTM path (per modality). Block owns 16 batch rows; h,c resident in LDS.
// Double-buffered x_t tile; loop-invariant Wih/Whh B-fragments and bias live
// in VGPRs; all A-fragments preloaded per step so DS latency overlaps WMMA.
// ---------------------------------------------------------------------------
template <int DIN, int KPAD>
__device__ void lstm_tile(const float* __restrict__ X, const __bf16* __restrict__ wih,
                          const __bf16* __restrict__ whh, const float* __restrict__ biasc,
                          const __bf16* __restrict__ wo, const float* __restrict__ bo,
                          float* __restrict__ xbuf, int modoff, char* smem)
{
    constexpr int NK = KPAD / 32;

    __bf16* xt    = (__bf16*)smem;                            // [2][16][KPAD]
    float*  gates = (float*)(smem + 2 * 16 * KPAD * 2);       // [16][132]
    float*  cst   = (float*)((char*)gates + 16 * 132 * 4);    // [16][32]
    __bf16* hst   = (__bf16*)((char*)cst + 16 * 32 * 4);      // [16][32]

    int tid = threadIdx.x, lane = tid & 31, wave = tid >> 5;
    int n0 = blockIdx.x * 16;
    int c0 = wave * 16;

    for (int i = tid; i < 2 * 16 * KPAD; i += 256) xt[i] = (__bf16)0.f;
    for (int i = tid; i < 512; i += 256) { cst[i] = 0.f; hst[i] = (__bf16)0.f; }
    __syncthreads();

    // Hoist loop-invariant weight fragments + bias into registers
    v16bf bwih[NK];
    #pragma unroll
    for (int ks = 0; ks < NK; ++ks)
        bwih[ks] = load_b16(wih, KPAD, c0, ks * 32, lane);
    v16bf bwhh = load_b16(whh, 32, c0, 0, lane);
    const float bias = biasc[c0 + (lane & 15)];
    const int   ncol = c0 + (lane & 15);
    const int   mb   = (lane >> 4) * 8;

    // static cell-update indices: elements (wave, lane) and (wave+8, lane)
    float*       g0 = gates + wave * 132 + lane;
    float*       g1 = gates + (wave + 8) * 132 + lane;
    const int    i0 = wave * 32 + lane;
    const int    i1 = (wave + 8) * 32 + lane;

    // preload t=0 into buffer 0
    {
        const float* src = X + (size_t)n0 * DIN;
        for (int i = tid; i < 16 * DIN; i += 256)
            xt[(i / DIN) * KPAD + (i % DIN)] = (__bf16)src[i];
    }
    __syncthreads();

    for (int t = 0; t < TT; ++t) {
        const __bf16* cur = xt + (t & 1) * 16 * KPAD;
        __bf16*       nxt = xt + ((t + 1) & 1) * 16 * KPAD;

        // stream t+1 tile while computing t (consumed after two barriers)
        if (t + 1 < TT) {
            const float* src = X + ((size_t)(t + 1) * NN + n0) * DIN;
            for (int i = tid; i < 16 * DIN; i += 256)
                nxt[(i / DIN) * KPAD + (i % DIN)] = (__bf16)src[i];
        }
        if (t + 2 < TT) {
            const float* pf = X + ((size_t)(t + 2) * NN + n0) * DIN;
            for (int i = tid * 32; i < 16 * DIN; i += 256 * 32)
                __builtin_prefetch(pf + i, 0, 1);   // global_prefetch_b8 -> L2 warm
        }

        // preload ALL A fragments, then run the dependent WMMA chain so the
        // ds_load latency overlaps the first matrix ops
        v16bf af[NK + 1];
        #pragma unroll
        for (int ks = 0; ks < NK; ++ks)
            af[ks] = load_a16(cur, KPAD, ks * 32, lane);
        af[NK] = load_a16(hst, 32, 0, lane);

        v8f acc = {};
        #pragma unroll
        for (int ks = 0; ks < NK; ++ks)
            acc = WMMA_BF16(af[ks], bwih[ks], acc);
        acc = WMMA_BF16(af[NK], bwhh, acc);

        #pragma unroll
        for (int r = 0; r < 8; ++r) gates[(mb + r) * 132 + ncol] = acc[r] + bias;
        __syncthreads();

        // cell update (gate order i,f,g,o); two static elements per thread,
        // interleaved to keep the trans pipe busy
        {
            float gi0 = g0[0],  gi1 = g1[0];
            float gf0 = g0[32], gf1 = g1[32];
            float gg0 = g0[64], gg1 = g1[64];
            float go0 = g0[96], go1 = g1[96];
            float c0v = sigm(gf0) * cst[i0] + sigm(gi0) * ftanh(gg0);
            float c1v = sigm(gf1) * cst[i1] + sigm(gi1) * ftanh(gg1);
            cst[i0] = c0v;
            cst[i1] = c1v;
            hst[i0] = (__bf16)(sigm(go0) * ftanh(c0v));
            hst[i1] = (__bf16)(sigm(go1) * ftanh(c1v));
        }
        __syncthreads();
    }

    // out = relu(h_T @ Wo^T + bo)
    if (wave < 2) {
        v8f acc = {};
        v16bf a = load_a16(hst, 32, 0, lane);
        v16bf b = load_b16(wo, 32, c0, 0, lane);
        acc = WMMA_BF16(a, b, acc);
        float ob = bo[ncol];
        #pragma unroll
        for (int r = 0; r < 8; ++r)
            xbuf[(size_t)(n0 + mb + r) * 96 + modoff + ncol] = fmaxf(acc[r] + ob, 0.f);
    }
}

// ---------------------------------------------------------------------------
// Fused modality-subnet kernel: gridDim.y selects {text, audio-LSTM, video-LSTM}
// so all three independent stages run concurrently across the device.
// ---------------------------------------------------------------------------
__global__ __launch_bounds__(256) void subnet_kernel(
    const float* __restrict__ l, const float* __restrict__ a, const float* __restrict__ v,
    const float* __restrict__ lb1, const float* __restrict__ lb2,
    const float* __restrict__ abo, const float* __restrict__ vbo,
    char* __restrict__ ws, float* __restrict__ xbuf)
{
    __shared__ __align__(16) char smem[SMEM_BYTES];

    if (blockIdx.y == 0) {
        text_tile(l, lb1, lb2, ws, xbuf, smem);
    } else if (blockIdx.y == 1) {
        lstm_tile<DAA, KAP>(a, (const __bf16*)(ws + OFF_AWIH), (const __bf16*)(ws + OFF_AWHH),
                            (const float*)(ws + OFF_ABIAS), (const __bf16*)(ws + OFF_AWO),
                            abo, xbuf, 32, smem);
    } else {
        lstm_tile<DVV, KVP>(v, (const __bf16*)(ws + OFF_VWIH), (const __bf16*)(ws + OFF_VWHH),
                            (const float*)(ws + OFF_VBIAS), (const __bf16*)(ws + OFF_VWO),
                            vbo, xbuf, 64, smem);
    }
}

// ---------------------------------------------------------------------------
// Per-sample transformer block + head. One wave (32 lanes) per sample.
// ---------------------------------------------------------------------------
__global__ __launch_bounds__(256) void fuse_kernel(
    const float* __restrict__ xbuf,
    const float* __restrict__ Wq, const float* __restrict__ bq,
    const float* __restrict__ Wk, const float* __restrict__ bk,
    const float* __restrict__ Wv, const float* __restrict__ bv,
    const float* __restrict__ Wz, const float* __restrict__ bz,
    const float* __restrict__ Wff, const float* __restrict__ bff,
    const float* __restrict__ g1, const float* __restrict__ be1,
    const float* __restrict__ g2, const float* __restrict__ be2,
    const float* __restrict__ Wp1, const float* __restrict__ bp1,
    const float* __restrict__ Wp2, const float* __restrict__ bp2,
    float* __restrict__ out)
{
    __shared__ float shz[8][3][32];
    __shared__ float shc[8][3][32];
    __shared__ float shn[8][3][32];
    __shared__ float shf[8][3][32];
    __shared__ float shh[8][32];

    int lane = threadIdx.x & 31, wave = threadIdx.x >> 5;
    int n = blockIdx.x * 8 + wave;
    const float* xr = xbuf + (size_t)n * 96;
    float x0 = xr[lane], x1 = xr[32 + lane], x2 = xr[64 + lane];

    // LayerNorm 1 over all 96 values
    float s = x0 + x1 + x2;
    for (int m = 1; m < 32; m <<= 1) s += __shfl_xor(s, m, 32);
    float mu = s * (1.f / 96.f);
    float d0 = x0 - mu, d1 = x1 - mu, d2 = x2 - mu;
    float vv = d0 * d0 + d1 * d1 + d2 * d2;
    for (int m = 1; m < 32; m <<= 1) vv += __shfl_xor(vv, m, 32);
    float inv = frsq(vv * (1.f / 96.f) + 1e-5f);
    shz[wave][0][lane] = d0 * inv * g1[lane]      + be1[lane];
    shz[wave][1][lane] = d1 * inv * g1[32 + lane] + be1[32 + lane];
    shz[wave][2][lane] = d2 * inv * g1[64 + lane] + be1[64 + lane];
    __syncthreads();

    // Q/K/V: lane = (head h, dim e)
    int hd = lane >> 3, e = lane & 7;
    const float* wq = Wq + (hd * 8 + e) * 32;
    const float* wk = Wk + (hd * 8 + e) * 32;
    const float* wv = Wv + (hd * 8 + e) * 32;
    float q[3] = { bq[hd * 8 + e], bq[hd * 8 + e], bq[hd * 8 + e] };
    float k[3] = { bk[hd * 8 + e], bk[hd * 8 + e], bk[hd * 8 + e] };
    float v[3] = { bv[hd * 8 + e], bv[hd * 8 + e], bv[hd * 8 + e] };
    for (int d = 0; d < 32; ++d) {
        float a0 = shz[wave][0][d], a1 = shz[wave][1][d], a2 = shz[wave][2][d];
        float cq = wq[d], ck = wk[d], cv = wv[d];
        q[0] += a0 * cq; q[1] += a1 * cq; q[2] += a2 * cq;
        k[0] += a0 * ck; k[1] += a1 * ck; k[2] += a2 * ck;
        v[0] += a0 * cv; v[1] += a1 * cv; v[2] += a2 * cv;
    }

    // scores per head: reduce over e (8-lane aligned group)
    float att[3][3];
    #pragma unroll
    for (int si = 0; si < 3; ++si)
        #pragma unroll
        for (int ti = 0; ti < 3; ++ti) {
            float p = q[si] * k[ti];
            p += __shfl_xor(p, 1, 32);
            p += __shfl_xor(p, 2, 32);
            p += __shfl_xor(p, 4, 32);
            att[si][ti] = p;
        }
    // softmax over t, THEN divide by sqrt(8) (faithful to source)
    const float isq = 0.3535533905932738f;
    #pragma unroll
    for (int si = 0; si < 3; ++si) {
        float mx = fmaxf(att[si][0], fmaxf(att[si][1], att[si][2]));
        float e0 = __expf(att[si][0] - mx), e1 = __expf(att[si][1] - mx), e2 = __expf(att[si][2] - mx);
        float is = isq * frcp(e0 + e1 + e2);
        att[si][0] = e0 * is; att[si][1] = e1 * is; att[si][2] = e2 * is;
    }
    // Zh, concat heads: column index == lane
    #pragma unroll
    for (int si = 0; si < 3; ++si)
        shc[wave][si][lane] = att[si][0] * v[0] + att[si][1] * v[1] + att[si][2] * v[2];
    __syncthreads();

    // z2 = Zc @ Wz^T + bz + x  (residual 1)
    float y0 = bz[lane] + x0, y1 = bz[lane] + x1, y2 = bz[lane] + x2;
    const float* wzr = Wz + lane * 32;
    for (int d = 0; d < 32; ++d) {
        float c = wzr[d];
        y0 += shc[wave][0][d] * c; y1 += shc[wave][1][d] * c; y2 += shc[wave][2][d] * c;
    }

    // LayerNorm 2
    float s2 = y0 + y1 + y2;
    for (int m = 1; m < 32; m <<= 1) s2 += __shfl_xor(s2, m, 32);
    float mu2 = s2 * (1.f / 96.f);
    float e0 = y0 - mu2, e1 = y1 - mu2, e2 = y2 - mu2;
    float vv2 = e0 * e0 + e1 * e1 + e2 * e2;
    for (int m = 1; m < 32; m <<= 1) vv2 += __shfl_xor(vv2, m, 32);
    float inv2 = frsq(vv2 * (1.f / 96.f) + 1e-5f);
    shn[wave][0][lane] = e0 * inv2 * g2[lane]      + be2[lane];
    shn[wave][1][lane] = e1 * inv2 * g2[32 + lane] + be2[32 + lane];
    shn[wave][2][lane] = e2 * inv2 * g2[64 + lane] + be2[64 + lane];
    __syncthreads();

    // Z = Zn @ Wff^T + bff + z2  (residual 2)
    float f0 = bff[lane] + y0, f1 = bff[lane] + y1, f2 = bff[lane] + y2;
    const float* wfr = Wff + lane * 32;
    for (int d = 0; d < 32; ++d) {
        float c = wfr[d];
        f0 += shn[wave][0][d] * c; f1 += shn[wave][1][d] * c; f2 += shn[wave][2][d] * c;
    }
    shf[wave][0][lane] = f0; shf[wave][1][lane] = f1; shf[wave][2][lane] = f2;
    __syncthreads();

    // head: h_j = relu(flat(96) . Wp1[j] + bp1[j]),  j = lane
    const float* zfl = &shf[wave][0][0];
    const float* wp  = Wp1 + lane * 96;
    float hh = bp1[lane];
    for (int i = 0; i < 96; ++i) hh += zfl[i] * wp[i];
    shh[wave][lane] = fmaxf(hh, 0.f);
    __syncthreads();

    if (lane == 0) {
        float u0 = bp2[0], u1 = bp2[1];
        for (int j = 0; j < 32; ++j) {
            float hv = shh[wave][j];
            u0 += hv * Wp2[j];
            u1 += hv * Wp2[32 + j];
        }
        float mx = fmaxf(u0, u1);
        float ea = __expf(u0 - mx), eb = __expf(u1 - mx);
        float is = frcp(ea + eb);
        out[(size_t)n * 2]     = ea * is;
        out[(size_t)n * 2 + 1] = eb * is;
    }
}

// ---------------------------------------------------------------------------
// Launch
// ---------------------------------------------------------------------------
extern "C" void kernel_launch(void* const* d_in, const int* in_sizes, int n_in,
                              void* d_out, int out_size, void* d_ws, size_t ws_size,
                              hipStream_t stream) {
    const float* l    = (const float*)d_in[0];
    const float* a    = (const float*)d_in[1];
    const float* v    = (const float*)d_in[2];
    const float* lW1  = (const float*)d_in[3];
    const float* lb1  = (const float*)d_in[4];
    const float* lW2  = (const float*)d_in[5];
    const float* lb2  = (const float*)d_in[6];
    const float* aWih = (const float*)d_in[7];
    const float* aWhh = (const float*)d_in[8];
    const float* abih = (const float*)d_in[9];
    const float* abhh = (const float*)d_in[10];
    const float* aWo  = (const float*)d_in[11];
    const float* abo  = (const float*)d_in[12];
    const float* vWih = (const float*)d_in[13];
    const float* vWhh = (const float*)d_in[14];
    const float* vbih = (const float*)d_in[15];
    const float* vbhh = (const float*)d_in[16];
    const float* vWo  = (const float*)d_in[17];
    const float* vbo  = (const float*)d_in[18];
    const float* Wq   = (const float*)d_in[19];
    const float* bq   = (const float*)d_in[20];
    const float* Wk   = (const float*)d_in[21];
    const float* bk   = (const float*)d_in[22];
    const float* Wv   = (const float*)d_in[23];
    const float* bv   = (const float*)d_in[24];
    const float* Wz   = (const float*)d_in[25];
    const float* bz   = (const float*)d_in[26];
    const float* Wff  = (const float*)d_in[27];
    const float* bff  = (const float*)d_in[28];
    const float* g1   = (const float*)d_in[29];
    const float* be1  = (const float*)d_in[30];
    const float* g2   = (const float*)d_in[31];
    const float* be2  = (const float*)d_in[32];
    const float* Wp1  = (const float*)d_in[33];
    const float* bp1  = (const float*)d_in[34];
    const float* Wp2  = (const float*)d_in[35];
    const float* bp2  = (const float*)d_in[36];

    char*  ws   = (char*)d_ws;
    float* xbuf = (float*)(ws + OFF_XBUF);
    float* out  = (float*)d_out;

    prep_kernel<<<32, 256, 0, stream>>>(lW1, lW2, aWih, aWhh, abih, abhh,
                                        vWih, vWhh, vbih, vbhh, aWo, vWo, ws);

    subnet_kernel<<<dim3(NN / 16, 3), 256, 0, stream>>>(l, a, v, lb1, lb2, abo, vbo, ws, xbuf);

    fuse_kernel<<<NN / 8, 256, 0, stream>>>(xbuf, Wq, bq, Wk, bk, Wv, bv, Wz, bz,
                                            Wff, bff, g1, be1, g2, be2,
                                            Wp1, bp1, Wp2, bp2, out);
}
